// TRANSFORMER_CONV_28887950033712
// MI455X (gfx1250) — matrix operand order repaired and checked by reading, not compiled, not run on hardware
//
#include <hip/hip_runtime.h>
#include <cstdint>

// TransformerConv x5 for MI455X (gfx1250, wave32).
// Dense projections run on v_wmma_f32_16x16x32_bf16 using split-bf16 (hi+lo)
// 3-term emulation for ~fp32 accuracy; q/k/v share one fused kernel so the A
// operand is loaded+converted once per tile. Edge phase is gather/atomic bound
// and recomputes edge_attr@We from LDS (staged via async-to-LDS DMA when the
// toolchain exposes the gfx1250 builtin) instead of materializing 410MB.

#define NNODES 50000
#define NEDGES 800000
#define ED 16
#define NH 4
#define CH 32
#define HC 128

typedef __attribute__((ext_vector_type(16))) __bf16 v16bf;
typedef __attribute__((ext_vector_type(8)))  float  v8f;
typedef int v4i __attribute__((vector_size(16)));

#if defined(__has_builtin)
#if __has_builtin(__builtin_amdgcn_global_load_async_to_lds_b128) && \
    __has_builtin(__builtin_amdgcn_s_wait_asynccnt)
#define HAVE_ASYNC_LDS 1
#endif
#endif
#ifndef HAVE_ASYNC_LDS
#define HAVE_ASYNC_LDS 0
#endif

// monotonic float <-> uint key for atomicMax on floats
__device__ __forceinline__ unsigned fkey(float x) {
  unsigned u = __float_as_uint(x);
  return (u & 0x80000000u) ? ~u : (u | 0x80000000u);
}
__device__ __forceinline__ float funkey(unsigned m) {
  unsigned u = (m & 0x80000000u) ? (m & 0x7FFFFFFFu) : ~m;
  return __uint_as_float(u);
}

// Stage We [16x128] fp32 (8KB) into LDS. Async DMA path when available
// (global_load_async_to_lds_b128, ASYNCcnt), else plain loads. blockDim==256.
__device__ __forceinline__ void stage_We(const float* __restrict__ We, float* sWe) {
#if HAVE_ASYNC_LDS
  {
    typedef __attribute__((address_space(1))) v4i* gv4p;   // global (AS1) int4*
    typedef __attribute__((address_space(3))) v4i* lv4p;   // LDS    (AS3) int4*
    __attribute__((address_space(1))) char* g1 =
        (__attribute__((address_space(1))) char*)(char*)We;
    __attribute__((address_space(3))) char* l3 =
        (__attribute__((address_space(3))) char*)sWe;
#pragma unroll
    for (int it = 0; it < (ED * HC * 4) / (256 * 16); ++it) {  // 2 iters: 512 x b128
      const int i = it * 256 + (int)threadIdx.x;
      __builtin_amdgcn_global_load_async_to_lds_b128(
          (gv4p)(g1 + (size_t)i * 16),
          (lv4p)(l3 + (size_t)i * 16),
          0, 0);
    }
    __builtin_amdgcn_s_wait_asynccnt(0);
  }
#else
  for (int i = threadIdx.x; i < ED * HC; i += blockDim.x) sWe[i] = We[i];
#endif
  __syncthreads();
}

// ---------------------------------------------------------------------------
// Weight prep: W [128 x Nout] fp32 row-major -> swizzled bf16 hi/lo so that in
// the GEMM each lane loads 16 contiguous bf16 per k-chunk.
// Layout index: (((nt*4 + kc)*32 + lane)*16 + j)
//   k = kc*32 + (lane>>4)*16 + j   (B operand: lanes 0-15 K=0..15, 16-31 K=16..31)
//   n = nt*16 + (lane&15)
// ---------------------------------------------------------------------------
__global__ void prep_b_kernel(const float* __restrict__ W, __bf16* __restrict__ Bh,
                              __bf16* __restrict__ Bl, int nTiles, int Nout)
{
  int idx = blockIdx.x * blockDim.x + threadIdx.x;
  int total = nTiles * 2048;              // 4 kchunks * 32 lanes * 16 elems
  if (idx >= total) return;
  int j    = idx & 15;
  int lane = (idx >> 4) & 31;
  int kc   = (idx >> 9) & 3;
  int nt   = idx >> 11;
  int k = kc * 32 + ((lane >> 4) << 4) + j;
  int n = nt * 16 + (lane & 15);
  float xv = W[(size_t)k * Nout + n];
  __bf16 hi = (__bf16)xv;
  Bh[idx] = hi;
  Bl[idx] = (__bf16)(xv - (float)hi);
}

// Build the A operand (split-bf16 hi/lo) for one 32-wide k-chunk.
// A operand layout used here (contiguous k, same order as prep_b_kernel's B):
//   lane l: m = l&15, khalf = l>>4; elem j -> K = khalf*16 + j
__device__ __forceinline__ void build_a(const float* __restrict__ arow, int kb, int khi,
                                        v16bf& a_hi, v16bf& a_lo)
{
#pragma unroll
  for (int j = 0; j < 16; ++j) {
    const int k = kb + (khi << 4) + j;
    float xv = arow[k];
    __bf16 hi = (__bf16)xv;
    a_hi[j] = hi;
    a_lo[j] = (__bf16)(xv - (float)hi);
  }
}

#define WMMA3(ACC, AH, AL, BH, BL)                                                              \
  ACC = __builtin_amdgcn_wmma_f32_16x16x32_bf16(false, AH, false, BH, (short)0, ACC, false, false); \
  ACC = __builtin_amdgcn_wmma_f32_16x16x32_bf16(false, AH, false, BL, (short)0, ACC, false, false); \
  ACC = __builtin_amdgcn_wmma_f32_16x16x32_bf16(false, AL, false, BH, (short)0, ACC, false, false)

// ---------------------------------------------------------------------------
// Fused q/k/v: three [M,128]x[128,128] GEMMs sharing one A fetch+convert.
// One wave per 16-row x 16-col tile triple; blockDim.x = 256 (8 column tiles).
// 36 WMMAs per wave (4 k-chunks x 3 split terms x 3 outputs).
// ---------------------------------------------------------------------------
__global__ void gemm128_qkv_wmma_kernel(const float* __restrict__ A,
    const __bf16* __restrict__ Bh0, const __bf16* __restrict__ Bl0,
    const __bf16* __restrict__ Bh1, const __bf16* __restrict__ Bl1,
    const __bf16* __restrict__ Bh2, const __bf16* __restrict__ Bl2,
    const float* __restrict__ b0, const float* __restrict__ b1, const float* __restrict__ b2,
    float* __restrict__ C0, float* __restrict__ C1, float* __restrict__ C2)
{
  const int lane = threadIdx.x & 31;
  const int nt   = threadIdx.x >> 5;
  const int m0   = blockIdx.x * 16;
  const int khi  = lane >> 4;

  const float* __restrict__ arow = A + (size_t)(m0 + (lane & 15)) * 128;
  v8f acc0 = {}, acc1 = {}, acc2 = {};

#pragma unroll
  for (int kc = 0; kc < 4; ++kc) {
    v16bf a_hi, a_lo;
    build_a(arow, kc * 32, khi, a_hi, a_lo);
    const size_t boff = (((size_t)nt * 4 + kc) * 32 + lane) * 16;
    {
      v16bf bh = *(const v16bf*)(Bh0 + boff);
      v16bf bl = *(const v16bf*)(Bl0 + boff);
      WMMA3(acc0, a_hi, a_lo, bh, bl);
    }
    {
      v16bf bh = *(const v16bf*)(Bh1 + boff);
      v16bf bl = *(const v16bf*)(Bl1 + boff);
      WMMA3(acc1, a_hi, a_lo, bh, bl);
    }
    {
      v16bf bh = *(const v16bf*)(Bh2 + boff);
      v16bf bl = *(const v16bf*)(Bl2 + boff);
      WMMA3(acc2, a_hi, a_lo, bh, bl);
    }
  }

  const int col = nt * 16 + (lane & 15);
  const float bv0 = b0[col], bv1 = b1[col], bv2 = b2[col];
#pragma unroll
  for (int r = 0; r < 8; ++r) {
    const size_t row = (size_t)(m0 + r + 8 * khi);   // C layout: VGPR r -> M = r + 8*khalf
    C0[row * HC + col] = acc0[r] + bv0;
    C1[row * HC + col] = acc1[r] + bv1;
    C2[row * HC + col] = acc2[r] + bv2;
  }
}

// Single GEMM (used for the skip projection): C[M x ldc] = A[M x 128]*W + bias.
// blockDim.x = nTiles*32; M multiple of 16 (50000 = 3125*16).
__global__ void gemm128_wmma_kernel(const float* __restrict__ A,
                                    const __bf16* __restrict__ Bh,
                                    const __bf16* __restrict__ Bl,
                                    const float* __restrict__ bias,
                                    float* __restrict__ C,
                                    int ldc)
{
  const int lane = threadIdx.x & 31;
  const int nt   = threadIdx.x >> 5;
  const int m0   = blockIdx.x * 16;
  const int khi  = lane >> 4;

  const float* __restrict__ arow = A + (size_t)(m0 + (lane & 15)) * 128;
  v8f acc = {};

#pragma unroll
  for (int kc = 0; kc < 4; ++kc) {
    v16bf a_hi, a_lo;
    build_a(arow, kc * 32, khi, a_hi, a_lo);
    const size_t boff = (((size_t)nt * 4 + kc) * 32 + lane) * 16;
    v16bf bh = *(const v16bf*)(Bh + boff);
    v16bf bl = *(const v16bf*)(Bl + boff);
    WMMA3(acc, a_hi, a_lo, bh, bl);
  }

  const int col = nt * 16 + (lane & 15);
  const float bv = bias[col];
#pragma unroll
  for (int r = 0; r < 8; ++r) {
    const int row = m0 + r + 8 * khi;
    C[(size_t)row * ldc + col] = acc[r] + bv;
  }
}

// ---------------------------------------------------------------------------
// Edge attention logits: alpha[e,h] = <q[dst], k[src] (+ edge_attr@We)> / sqrt(C)
// One wave per edge; lane owns float4 (head h = lane>>3); shfl reduce per head.
// Also folds the segment-max via atomicMax on monotonic uint keys.
// ---------------------------------------------------------------------------
template<bool HASEDGE>
__global__ void edge_alpha_kernel(const int* __restrict__ srcA, const int* __restrict__ dstA,
                                  const float* __restrict__ q, const float* __restrict__ k,
                                  const float* __restrict__ eattr, const float* __restrict__ We,
                                  float* __restrict__ alpha, unsigned* __restrict__ amax)
{
  __shared__ float sWe[HASEDGE ? (ED * HC) : 1];
  if constexpr (HASEDGE) stage_We(We, sWe);

  const int lane = threadIdx.x & 31;
  const int e = blockIdx.x * (blockDim.x >> 5) + (threadIdx.x >> 5);
  if (e >= NEDGES) return;
  const int s = srcA[e];
  const int d = dstA[e];
  float4 qv = ((const float4*)(q + (size_t)d * HC))[lane];
  float4 kv = ((const float4*)(k + (size_t)s * HC))[lane];
  if constexpr (HASEDGE) {
    const float* ea = eattr + (size_t)e * ED;
    float e0 = 0.f, e1 = 0.f, e2 = 0.f, e3 = 0.f;
#pragma unroll
    for (int t = 0; t < ED; ++t) {
      const float a = ea[t];
      const float* wr = &sWe[t * HC + lane * 4];
      e0 += a * wr[0]; e1 += a * wr[1]; e2 += a * wr[2]; e3 += a * wr[3];
    }
    kv.x += e0; kv.y += e1; kv.z += e2; kv.w += e3;
  }
  float p = qv.x * kv.x + qv.y * kv.y + qv.z * kv.z + qv.w * kv.w;
  p += __shfl_xor(p, 1, 32);
  p += __shfl_xor(p, 2, 32);
  p += __shfl_xor(p, 4, 32);
  if ((lane & 7) == 0) {
    const int h = lane >> 3;
    const float a = p * 0.17677669529663688f;   // 1/sqrt(32)
    alpha[(size_t)e * NH + h] = a;
    atomicMax(&amax[(size_t)d * NH + h], fkey(a));
  }
}

// alpha[e,h] = exp(alpha - amax[dst,h]); denom[dst,h] += alpha
__global__ void edge_exp_kernel(float* __restrict__ alpha, const unsigned* __restrict__ amax,
                                const int* __restrict__ dstA, float* __restrict__ denom)
{
  const int i = blockIdx.x * blockDim.x + threadIdx.x;
  if (i >= NEDGES * NH) return;
  const int e = i >> 2, h = i & 3;
  const int d = dstA[e];
  const float m = funkey(amax[(size_t)d * NH + h]);
  const float a = __expf(alpha[i] - m);
  alpha[i] = a;
  atomicAdd(&denom[(size_t)d * NH + h], a);
}

// agg[dst] += (v[src] (+ee)) * alpha/(denom+1e-16), normalization folded in.
template<bool HASEDGE>
__global__ void edge_aggr_kernel(const int* __restrict__ srcA, const int* __restrict__ dstA,
                                 const float* __restrict__ v, const float* __restrict__ alpha,
                                 const float* __restrict__ denom,
                                 const float* __restrict__ eattr, const float* __restrict__ We,
                                 float* __restrict__ outAgg)
{
  __shared__ float sWe[HASEDGE ? (ED * HC) : 1];
  if constexpr (HASEDGE) stage_We(We, sWe);

  const int lane = threadIdx.x & 31;
  const int e = blockIdx.x * (blockDim.x >> 5) + (threadIdx.x >> 5);
  if (e >= NEDGES) return;
  const int s = srcA[e];
  const int d = dstA[e];
  float4 vv = ((const float4*)(v + (size_t)s * HC))[lane];
  if constexpr (HASEDGE) {
    const float* ea = eattr + (size_t)e * ED;
    float e0 = 0.f, e1 = 0.f, e2 = 0.f, e3 = 0.f;
#pragma unroll
    for (int t = 0; t < ED; ++t) {
      const float a = ea[t];
      const float* wr = &sWe[t * HC + lane * 4];
      e0 += a * wr[0]; e1 += a * wr[1]; e2 += a * wr[2]; e3 += a * wr[3];
    }
    vv.x += e0; vv.y += e1; vv.z += e2; vv.w += e3;
  }
  const int h = lane >> 3;
  const float wgt = alpha[(size_t)e * NH + h] / (denom[(size_t)d * NH + h] + 1e-16f);
  float* op = outAgg + (size_t)d * HC + lane * 4;
  atomicAdd(op + 0, vv.x * wgt);
  atomicAdd(op + 1, vv.y * wgt);
  atomicAdd(op + 2, vv.z * wgt);
  atomicAdd(op + 3, vv.w * wgt);
}

// beta-gated skip + ReLU, concat layers (dout = 128). One wave per node.
__global__ void node_finish_kernel(const float* __restrict__ agg, const float* __restrict__ xr,
                                   const float* __restrict__ Wbeta, float* __restrict__ hnext)
{
  const int lane = threadIdx.x & 31;
  const int n = blockIdx.x * (blockDim.x >> 5) + (threadIdx.x >> 5);
  if (n >= NNODES) return;
  float4 o  = ((const float4*)(agg + (size_t)n * HC))[lane];
  float4 r  = ((const float4*)(xr  + (size_t)n * HC))[lane];
  float4 w0 = ((const float4*)(Wbeta))[lane];
  float4 w1 = ((const float4*)(Wbeta + HC))[lane];
  float4 w2 = ((const float4*)(Wbeta + 2 * HC))[lane];
  float p = o.x*w0.x + o.y*w0.y + o.z*w0.z + o.w*w0.w
          + r.x*w1.x + r.y*w1.y + r.z*w1.z + r.w*w1.w
          + (o.x-r.x)*w2.x + (o.y-r.y)*w2.y + (o.z-r.z)*w2.z + (o.w-r.w)*w2.w;
  p += __shfl_xor(p, 16, 32);
  p += __shfl_xor(p, 8, 32);
  p += __shfl_xor(p, 4, 32);
  p += __shfl_xor(p, 2, 32);
  p += __shfl_xor(p, 1, 32);
  const float beta = 1.f / (1.f + __expf(-p));
  float4 y;
  y.x = fmaxf(beta * r.x + (1.f - beta) * o.x, 0.f);
  y.y = fmaxf(beta * r.y + (1.f - beta) * o.y, 0.f);
  y.z = fmaxf(beta * r.z + (1.f - beta) * o.z, 0.f);
  y.w = fmaxf(beta * r.w + (1.f - beta) * o.w, 0.f);
  ((float4*)(hnext + (size_t)n * HC))[lane] = y;
}

// Last layer: concat=False -> mean over heads, dout = 32. One wave per node.
__global__ void node_finish_mean_kernel(const float* __restrict__ agg, const float* __restrict__ xr,
                                        const float* __restrict__ Wbeta, float* __restrict__ out)
{
  const int lane = threadIdx.x & 31;   // channel c
  const int n = blockIdx.x * (blockDim.x >> 5) + (threadIdx.x >> 5);
  if (n >= NNODES) return;
  const float* ar = agg + (size_t)n * HC;
  const float o = 0.25f * (ar[lane] + ar[32 + lane] + ar[64 + lane] + ar[96 + lane]);
  const float r = xr[(size_t)n * CH + lane];
  float p = o * Wbeta[lane] + r * Wbeta[CH + lane] + (o - r) * Wbeta[2 * CH + lane];
  p += __shfl_xor(p, 16, 32);
  p += __shfl_xor(p, 8, 32);
  p += __shfl_xor(p, 4, 32);
  p += __shfl_xor(p, 2, 32);
  p += __shfl_xor(p, 1, 32);
  const float beta = 1.f / (1.f + __expf(-p));
  out[(size_t)n * CH + lane] = fmaxf(beta * r + (1.f - beta) * o, 0.f);
}

// ---------------------------------------------------------------------------
struct LayerP {
  const float *Wbeta, *We, *Wk, *Wq, *Wskip, *Wv, *bk, *bq, *bskip, *bv;
};

extern "C" void kernel_launch(void* const* d_in, const int* in_sizes, int n_in,
                              void* d_out, int out_size, void* d_ws, size_t ws_size,
                              hipStream_t stream)
{
  (void)in_sizes; (void)out_size; (void)ws_size;
  if (n_in < 51) return;   // 3 tensors + 10+10+10+9+9 param leaves

  const float* x     = (const float*)d_in[0];
  const int*   ei    = (const int*)d_in[1];     // edge_index [2,E]
  const float* eattr = (const float*)d_in[2];
  const int*   srcA  = ei;
  const int*   dstA  = ei + NEDGES;

  // Param leaves: jax pytree order = tuple order, dict keys sorted
  // (Wbeta, We?, Wk, Wq, Wskip, Wv, bk, bq, bskip, bv), None (We) dropped.
  LayerP L[5];
  int idx = 3;
  for (int l = 0; l < 5; ++l) {
    const bool hasE = (l < 3);
    L[l].Wbeta = (const float*)d_in[idx++];
    L[l].We    = hasE ? (const float*)d_in[idx++] : nullptr;
    L[l].Wk    = (const float*)d_in[idx++];
    L[l].Wq    = (const float*)d_in[idx++];
    L[l].Wskip = (const float*)d_in[idx++];
    L[l].Wv    = (const float*)d_in[idx++];
    L[l].bk    = (const float*)d_in[idx++];
    L[l].bq    = (const float*)d_in[idx++];
    L[l].bskip = (const float*)d_in[idx++];
    L[l].bv    = (const float*)d_in[idx++];
  }

  // Workspace carve (~194 MB), 256B aligned slices.
  char* wsp = (char*)d_ws;
  auto carve = [&](size_t bytes) -> void* {
    void* p = (void*)wsp;
    wsp += (bytes + 255) & ~(size_t)255;
    return p;
  };
  float*    hA    = (float*)carve((size_t)NNODES * HC * 4);
  float*    hB    = (float*)carve((size_t)NNODES * HC * 4);
  float*    qb    = (float*)carve((size_t)NNODES * HC * 4);
  float*    kb    = (float*)carve((size_t)NNODES * HC * 4);
  float*    vb    = (float*)carve((size_t)NNODES * HC * 4);
  float*    xrb   = (float*)carve((size_t)NNODES * HC * 4);
  float*    agg   = (float*)carve((size_t)NNODES * HC * 4);
  float*    alpha = (float*)carve((size_t)NEDGES * NH * 4);
  unsigned* amax  = (unsigned*)carve((size_t)NNODES * NH * 4);
  float*    denom = (float*)carve((size_t)NNODES * NH * 4);

  __bf16* swzHi[5][4];
  __bf16* swzLo[5][4];
  const float* Wsrc[5][4];
  int wNT[5][4];
  for (int l = 0; l < 5; ++l) {
    Wsrc[l][0] = L[l].Wq; Wsrc[l][1] = L[l].Wk; Wsrc[l][2] = L[l].Wv; Wsrc[l][3] = L[l].Wskip;
    for (int t = 0; t < 4; ++t) {
      const int nT = (l == 4 && t == 3) ? 2 : 8;   // last-layer Wskip is 128x32
      wNT[l][t] = nT;
      const size_t elems = (size_t)nT * 2048;
      swzHi[l][t] = (__bf16*)carve(elems * 2);
      swzLo[l][t] = (__bf16*)carve(elems * 2);
    }
  }

  // One-time (per call) weight swizzle into split-bf16 hi/lo.
  for (int l = 0; l < 5; ++l)
    for (int t = 0; t < 4; ++t) {
      const int nT = wNT[l][t];
      const int total = nT * 2048;
      prep_b_kernel<<<(total + 255) / 256, 256, 0, stream>>>(Wsrc[l][t], swzHi[l][t], swzLo[l][t],
                                                             nT, nT * 16);
    }

  const int mBlocks = NNODES / 16;     // 3125 (exact)
  const int eBlocks = NEDGES / 8;      // 8 waves per block
  const int nBlocks = (NNODES + 7) / 8;

  const float* hcur = x;
  float* hpp[2] = { hA, hB };
  for (int l = 0; l < 5; ++l) {
    const bool hasE = (l < 3);
    const bool last = (l == 4);

    (void)hipMemsetAsync(agg,   0, (size_t)NNODES * HC * 4, stream);
    (void)hipMemsetAsync(amax,  0, (size_t)NNODES * NH * 4, stream);   // 0 = lowest fkey
    (void)hipMemsetAsync(denom, 0, (size_t)NNODES * NH * 4, stream);

    // Fused q/k/v projections: one A fetch+convert feeds 3 WMMA chains.
    gemm128_qkv_wmma_kernel<<<mBlocks, 256, 0, stream>>>(hcur,
        swzHi[l][0], swzLo[l][0], swzHi[l][1], swzLo[l][1], swzHi[l][2], swzLo[l][2],
        L[l].bq, L[l].bk, L[l].bv, qb, kb, vb);
    const int skipNT = wNT[l][3];
    gemm128_wmma_kernel<<<mBlocks, skipNT * 32, 0, stream>>>(hcur, swzHi[l][3], swzLo[l][3],
                                                             L[l].bskip, xrb, skipNT * 16);

    if (hasE) edge_alpha_kernel<true ><<<eBlocks, 256, 0, stream>>>(srcA, dstA, qb, kb, eattr, L[l].We, alpha, amax);
    else      edge_alpha_kernel<false><<<eBlocks, 256, 0, stream>>>(srcA, dstA, qb, kb, nullptr, nullptr, alpha, amax);

    edge_exp_kernel<<<(NEDGES * NH + 255) / 256, 256, 0, stream>>>(alpha, amax, dstA, denom);

    if (hasE) edge_aggr_kernel<true ><<<eBlocks, 256, 0, stream>>>(srcA, dstA, vb, alpha, denom, eattr, L[l].We, agg);
    else      edge_aggr_kernel<false><<<eBlocks, 256, 0, stream>>>(srcA, dstA, vb, alpha, denom, nullptr, nullptr, agg);

    if (!last) {
      node_finish_kernel<<<nBlocks, 256, 0, stream>>>(agg, xrb, L[l].Wbeta, hpp[l & 1]);
      hcur = hpp[l & 1];
    } else {
      node_finish_mean_kernel<<<nBlocks, 256, 0, stream>>>(agg, xrb, L[l].Wbeta, (float*)d_out);
    }
  }
}
